// Decoder_64905545777804
// MI455X (gfx1250) — matrix operand order repaired and checked by reading, not compiled, not run on hardware
//
#include <hip/hip_runtime.h>
#include <hip/hip_bf16.h>

typedef __bf16 bf16_t;
typedef __attribute__((ext_vector_type(16))) __bf16 v16bf;
typedef __attribute__((ext_vector_type(8)))  __bf16 v8bf;
typedef __attribute__((ext_vector_type(4)))  __bf16 v4bf;
typedef __attribute__((ext_vector_type(8)))  float  v8f;

static __device__ __forceinline__ bf16_t f2bf(float f) { return (bf16_t)f; }
static __device__ __forceinline__ float  bf2f(bf16_t h) { return (float)h; }

static __device__ __forceinline__ v8f vzero() {
  v8f z;
#pragma unroll
  for (int i = 0; i < 8; ++i) z[i] = 0.f;
  return z;
}

// A operand (M=16 x K=32 bf16), W row-major (rows = out-channel, rs = row stride elems).
// lane<16 : row o0+lane, K k0+0..7 and k0+16..23
// lane>=16: row o0+lane-16, K k0+8..15 and k0+24..31
static __device__ __forceinline__ v16bf load_a16(const bf16_t* W, int rs, int o0, int k0, int lane) {
  const int r = lane & 15, hi = lane >> 4;
  const bf16_t* p = W + (size_t)(o0 + r) * rs + k0 + hi * 8;
  v8bf g0 = *(const v8bf*)p;
  v8bf g1 = *(const v8bf*)(p + 16);
  v16bf a;
#pragma unroll
  for (int i = 0; i < 8; ++i) { a[i] = g0[i]; a[8 + i] = g1[i]; }
  return a;
}

// B operand (K=32 x N=16 bf16), X pixel-major (rows = pixel/N, K contiguous).
// lane<16 : col n0+lane, K k0+0..7 and k0+16..23 ; lane>=16: col n0+lane-16, K k0+8..15 and k0+24..31
static __device__ __forceinline__ v16bf load_b16(const bf16_t* X, int rs, int n0, int k0, int lane) {
  const int r = lane & 15, hi = lane >> 4;
  const bf16_t* p = X + (size_t)(n0 + r) * rs + k0 + hi * 8;
  v8bf g0 = *(const v8bf*)p;
  v8bf g1 = *(const v8bf*)(p + 16);
  v16bf a;
#pragma unroll
  for (int i = 0; i < 8; ++i) { a[i] = g0[i]; a[8 + i] = g1[i]; }
  return a;
}

static __device__ __forceinline__ v8f wmma_bf16(v16bf a, v16bf b, v8f c) {
  return __builtin_amdgcn_wmma_f32_16x16x32_bf16(false, a, false, b, (short)0, c, false, false);
}

// ---------------- hypernetwork layer 1 (small, latency-trivial) ----------------
__global__ void k_hyper1(const float* __restrict__ lat,
                         const float* __restrict__ w1, const float* __restrict__ b1,
                         const float* __restrict__ cw1, const float* __restrict__ cb1,
                         float* __restrict__ h1, float* __restrict__ h2) {
  int idx = blockIdx.x * 256 + threadIdx.x;
  if (idx < 16 * 512) {
    int b = idx >> 9, n = idx & 511;
    float acc = b1[n];
    for (int k = 0; k < 512; ++k) acc = fmaf(lat[b * 512 + k], w1[(size_t)k * 512 + n], acc);
    h1[idx] = fmaxf(acc, 0.f);
  } else if (idx < 16 * 512 + 16 * 1024) {
    int j = idx - 16 * 512;
    int b = j >> 10, n = j & 1023;
    float acc = cb1[n];
    for (int k = 0; k < 512; ++k) acc = fmaf(lat[b * 512 + k], cw1[(size_t)k * 1024 + n], acc);
    h2[j] = fmaxf(acc, 0.f);
  }
}

// ---------------- hypernetwork layer 2 (HBM bound: streams wt_w2 134MB) ----------------
__global__ void k_hyper2(const float* __restrict__ h1, const float* __restrict__ w2,
                         const float* __restrict__ b2,
                         const float* __restrict__ h2, const float* __restrict__ cw2,
                         const float* __restrict__ cb2,
                         float* __restrict__ pw, float* __restrict__ p) {
  int n = blockIdx.x * 256 + threadIdx.x;
  if (n < 65568) {
    float acc[16];
#pragma unroll
    for (int b = 0; b < 16; ++b) acc[b] = b2[n];
    for (int k = 0; k < 512; ++k) {
      float w = w2[(size_t)k * 65568 + n];
#pragma unroll
      for (int b = 0; b < 16; ++b) acc[b] = fmaf(h1[b * 512 + k], w, acc[b]);
    }
#pragma unroll
    for (int b = 0; b < 16; ++b) pw[(size_t)b * 65568 + n] = acc[b];
  } else if (n < 65568 + 24832) {
    int m = n - 65568;
    float acc[16];
#pragma unroll
    for (int b = 0; b < 16; ++b) acc[b] = cb2[m];
    for (int k = 0; k < 1024; ++k) {
      float w = cw2[(size_t)k * 24832 + m];
#pragma unroll
      for (int b = 0; b < 16; ++b) acc[b] = fmaf(h2[b * 1024 + k], w, acc[b]);
    }
#pragma unroll
    for (int b = 0; b < 16; ++b) p[(size_t)b * 24832 + m] = acc[b];
  }
}

// ---------------- unpack dynamic params to bf16 weights / f32 biases ----------------
__global__ void k_unpack_wave(const float* __restrict__ pw, bf16_t* __restrict__ ww,
                              float* __restrict__ wbb) {
  int idx = blockIdx.x * 256 + threadIdx.x;
  if (idx >= 16 * 65568) return;
  int b = idx / 65568, j = idx % 65568;
  float v = pw[idx];
  if (j < 65536) ww[(size_t)b * 65536 + j] = f2bf(v * 0.02209708691207961f); // 1/sqrt(2048)
  else wbb[b * 32 + (j - 65536)] = v;
}

__global__ void k_unpack_c2c(const float* __restrict__ p,
                             bf16_t* __restrict__ w_in, float* __restrict__ b_in,
                             bf16_t* __restrict__ w_mid, float* __restrict__ b_mid,
                             bf16_t* __restrict__ w_out, float* __restrict__ b_out,
                             bf16_t* __restrict__ w_sk, float* __restrict__ b_sk) {
  int idx = blockIdx.x * 256 + threadIdx.x;
  if (idx >= 16 * 24832) return;
  int b = idx / 24832, j = idx % 24832;
  const float rs128 = 0.08838834764831845f; // 1/sqrt(128)
  const float rs64  = 0.125f;
  float v = p[idx];
  if (j < 8192)       w_in[(size_t)b * 8192 + j] = f2bf(v * rs128);
  else if (j < 8256)  b_in[b * 64 + j - 8192] = v;
  else if (j < 12352) w_mid[(size_t)b * 4096 + j - 8256] = f2bf(v * rs64);
  else if (j < 12416) b_mid[b * 64 + j - 12352] = v;
  else if (j < 16512) w_out[(size_t)b * 4096 + j - 12416] = f2bf(v * rs64);
  else if (j < 16576) b_out[b * 64 + j - 16512] = v;
  else if (j < 24768) w_sk[(size_t)b * 8192 + j - 16576] = f2bf(v * rs128);
  else                b_sk[b * 64 + j - 24768] = v;
}

// ---------------- transpose wave_bias (2048ch x 16384pix f32) -> (pix, ch) bf16 ----------------
__global__ void k_transpose_wb(const float* __restrict__ wb, bf16_t* __restrict__ wbT) {
  __shared__ float t[32][33];
  const int pt = blockIdx.x * 32, ct = blockIdx.y * 32;
  const int r = threadIdx.x >> 5, c = threadIdx.x & 31;
#pragma unroll
  for (int i = 0; i < 4; ++i) {
    int cl = r + i * 8;
    t[cl][c] = wb[(size_t)(ct + cl) * 16384 + pt + c];
  }
  __syncthreads();
#pragma unroll
  for (int i = 0; i < 4; ++i) {
    int pl = r + i * 8;
    wbT[(size_t)(pt + pl) * 2048 + ct + c] = f2bf(t[c][pl]);
  }
}

// ---------------- wave GEMM: (32x2048)@(2048x16384) per batch, +bias, cos/sin, stats ----------------
__global__ void __launch_bounds__(256) k_wavegemm(
    const bf16_t* __restrict__ wbT, const bf16_t* __restrict__ ww,
    const float* __restrict__ wbb, bf16_t* __restrict__ xin,
    float* __restrict__ wstats) {
  __shared__ float ssum[64];
  __shared__ float ssq[64];
  __shared__ bf16_t tile[8][16][72];
  const int b = blockIdx.y;
  const int wv = threadIdx.x >> 5;
  const int lane = threadIdx.x & 31;
  const int px = lane & 15;
  const int rhi = (lane >> 4) * 8;
  const int p0 = blockIdx.x * 128 + wv * 16;
  if (threadIdx.x < 64) { ssum[threadIdx.x] = 0.f; ssq[threadIdx.x] = 0.f; }
  __syncthreads();
  const bf16_t* Wb = ww + (size_t)b * 65536;
  v8f acc0 = vzero(), acc1 = vzero();
#pragma unroll 4
  for (int ks = 0; ks < 64; ++ks) {
    const int k0 = ks * 32;
    v16bf Bv = load_b16(wbT, 2048, p0, k0, lane);
    v16bf A0 = load_a16(Wb, 2048, 0, k0, lane);
    acc0 = wmma_bf16(A0, Bv, acc0);
    v16bf A1 = load_a16(Wb, 2048, 16, k0, lane);
    acc1 = wmma_bf16(A1, Bv, acc1);
  }
#pragma unroll
  for (int m = 0; m < 2; ++m) {
    v8f& a = m ? acc1 : acc0;
#pragma unroll
    for (int r = 0; r < 8; ++r) {
      int ch = m * 16 + rhi + r;                 // 0..31
      float v = a[r] + wbb[b * 32 + ch];
      float cv = __cosf(v), sv = __sinf(v);      // hw v_cos_f32 / v_sin_f32
      tile[wv][px][ch] = f2bf(cv);
      tile[wv][px][32 + ch] = f2bf(sv);
      atomicAdd(&ssum[ch], cv);       atomicAdd(&ssq[ch], cv * cv);
      atomicAdd(&ssum[32 + ch], sv);  atomicAdd(&ssq[32 + ch], sv * sv);
    }
  }
  { // coalesced copy of 64 wave channels per pixel into xin (ch 64..127)
    const int co = (lane >> 4) * 32;
    const bf16_t* srcp = &tile[wv][px][co];
    bf16_t* dst = xin + ((size_t)(b * 16384 + p0 + px)) * 128 + 64 + co;
#pragma unroll
    for (int i = 0; i < 4; ++i) *(v8bf*)(dst + i * 8) = *(const v8bf*)(srcp + i * 8);
  }
  __syncthreads();
  if (threadIdx.x < 64) {
    atomicAdd(&wstats[b * 128 + threadIdx.x], ssum[threadIdx.x]);
    atomicAdd(&wstats[b * 128 + 64 + threadIdx.x], ssq[threadIdx.x]);
  }
}

// ---------------- pointwise instance-norm on wave channels of xin (in place) ----------------
__global__ void k_norm(bf16_t* __restrict__ xin, const float* __restrict__ stats, int chbase) {
  int idx = blockIdx.x * 256 + threadIdx.x;
  int gi = idx & 7;
  int pix = (idx >> 3) & 16383;
  int b = idx >> 17;
  if (b >= 16) return;
  bf16_t* rp = xin + ((size_t)(b * 16384 + pix)) * 128 + chbase + gi * 8;
  v8bf v = *(v8bf*)rp;
  const float inv = 1.f / 16384.f;
#pragma unroll
  for (int i = 0; i < 8; ++i) {
    int ch = gi * 8 + i;
    float sm = stats[b * 128 + ch] * inv;
    float sq = stats[b * 128 + 64 + ch] * inv;
    float rs = rsqrtf(sq - sm * sm + 1e-5f);
    v[i] = f2bf((bf2f(v[i]) - sm) * rs);
  }
  *(v8bf*)rp = v;
}

// ---------------- depthwise sobel (ky/kx) + per-(b,ch) stats; channel-major bf16 out ----------------
__global__ void __launch_bounds__(256) k_sobel(const float* __restrict__ src,
                                               bf16_t* __restrict__ graw,
                                               float* __restrict__ gstats) {
  const int b = blockIdx.z, c = blockIdx.y;
  const int pix = blockIdx.x * 256 + threadIdx.x;
  const int y = pix >> 7, x = pix & 127;
  const float* im = src + ((size_t)b * 32 + c) * 16384;
  const float hp = 1.57079632679489662f;
  const float sn = __sinf(hp), cs = __cosf(hp);
  float v[3][3];
#pragma unroll
  for (int dy = 0; dy < 3; ++dy)
#pragma unroll
    for (int dx = 0; dx < 3; ++dx) {
      int yy = y + dy - 1, xx = x + dx - 1;
      v[dy][dx] = (yy >= 0 && yy < 128 && xx >= 0 && xx < 128) ? im[yy * 128 + xx] : 0.f;
    }
  float gy = sn * ((cs * (v[2][0] + v[2][2]) + v[2][1]) - (cs * (v[0][0] + v[0][2]) + v[0][1]));
  float gx = sn * ((cs * (v[0][2] + v[2][2]) + v[1][2]) - (cs * (v[0][0] + v[2][0]) + v[1][0]));
  graw[((size_t)b * 64 + 2 * c) * 16384 + pix] = f2bf(gy);
  graw[((size_t)b * 64 + 2 * c + 1) * 16384 + pix] = f2bf(gx);
  __shared__ float r0[256], r1[256], r2[256], r3[256];
  const int t = threadIdx.x;
  r0[t] = gy; r1[t] = gy * gy; r2[t] = gx; r3[t] = gx * gx;
  __syncthreads();
  for (int s = 128; s > 0; s >>= 1) {
    if (t < s) { r0[t] += r0[t + s]; r1[t] += r1[t + s]; r2[t] += r2[t + s]; r3[t] += r3[t + s]; }
    __syncthreads();
  }
  if (t == 0) {
    atomicAdd(&gstats[b * 128 + 2 * c], r0[0]);
    atomicAdd(&gstats[b * 128 + 64 + 2 * c], r1[0]);
    atomicAdd(&gstats[b * 128 + 2 * c + 1], r2[0]);
    atomicAdd(&gstats[b * 128 + 64 + 2 * c + 1], r3[0]);
  }
}

// ---------------- normalize g and transpose into xin channels 0..63 (pixel-major) ----------------
__global__ void k_packg(const bf16_t* __restrict__ graw, const float* __restrict__ gstats,
                        bf16_t* __restrict__ xin) {
  __shared__ bf16_t t[32][33];
  const int b = blockIdx.z;
  const int cb = blockIdx.y * 32;
  const int pb = blockIdx.x * 32;
  const int r = threadIdx.x >> 5, cidx = threadIdx.x & 31;
  const float inv = 1.f / 16384.f;
#pragma unroll
  for (int i = 0; i < 4; ++i) {
    int cl = r + i * 8;
    int ch = cb + cl;
    float sm = gstats[b * 128 + ch] * inv;
    float sq = gstats[b * 128 + 64 + ch] * inv;
    float rs = rsqrtf(sq - sm * sm + 1e-5f);
    float v = bf2f(graw[((size_t)b * 64 + ch) * 16384 + pb + cidx]);
    t[cl][cidx] = f2bf((v - sm) * rs);
  }
  __syncthreads();
  const int pl = threadIdx.x >> 3, gg = threadIdx.x & 7;
  bf16_t* dst = xin + ((size_t)(b * 16384) + pb + pl) * 128 + cb + gg * 4;
  v4bf o;
#pragma unroll
  for (int i = 0; i < 4; ++i) o[i] = t[gg * 4 + i][pl];
  *(v4bf*)dst = o;
}

// ---------------- per-step WMMA chain: in->mid->out (+skip), gated update ----------------
__global__ void __launch_bounds__(256) k_chain(
    const bf16_t* __restrict__ xin,
    const bf16_t* __restrict__ w_in, const float* __restrict__ b_in,
    const bf16_t* __restrict__ w_mid, const float* __restrict__ b_mid,
    const bf16_t* __restrict__ w_out, const float* __restrict__ b_out,
    const bf16_t* __restrict__ w_sk, const float* __restrict__ b_sk,
    const float* __restrict__ leakp,
    const float* __restrict__ prev, float* __restrict__ nxt) {
  __shared__ bf16_t tile[8][16][72];
  const int b = blockIdx.y;
  const int wv = threadIdx.x >> 5;
  const int lane = threadIdx.x & 31;
  const int px = lane & 15;
  const int rhi = (lane >> 4) * 8;
  const int p0 = blockIdx.x * 128 + wv * 16;

  const bf16_t* Xb = xin + (size_t)b * 16384 * 128;
  const bf16_t* Win = w_in + (size_t)b * 8192;
  const bf16_t* Wmid = w_mid + (size_t)b * 4096;
  const bf16_t* Wout = w_out + (size_t)b * 4096;
  const bf16_t* Wsk = w_sk + (size_t)b * 8192;
  const float* Bin = b_in + b * 64;
  const float* Bmid = b_mid + b * 64;
  const float* Bout = b_out + b * 64;
  const float* Bsk = b_sk + b * 64;

  v8f accI[4], accS[4];
#pragma unroll
  for (int m = 0; m < 4; ++m) { accI[m] = vzero(); accS[m] = vzero(); }

  // layer 1 (K=128) + skip path share B operand
#pragma unroll
  for (int ks = 0; ks < 4; ++ks) {
    const int k0 = ks * 32;
    v16bf Bv = load_b16(Xb, 128, p0, k0, lane);
#pragma unroll
    for (int m = 0; m < 4; ++m) {
      v16bf Aa = load_a16(Win, 128, m * 16, k0, lane);
      accI[m] = wmma_bf16(Aa, Bv, accI[m]);
      v16bf As = load_a16(Wsk, 128, m * 16, k0, lane);
      accS[m] = wmma_bf16(As, Bv, accS[m]);
    }
  }
  // bias + leaky relu, retile t1 (C-layout -> pixel-major bf16 in LDS)
#pragma unroll
  for (int m = 0; m < 4; ++m)
#pragma unroll
    for (int r = 0; r < 8; ++r) {
      int ch = m * 16 + rhi + r;
      float v = accI[m][r] + Bin[ch];
      v = v > 0.f ? v : 0.2f * v;
      tile[wv][px][ch] = f2bf(v);
    }
  // layer 2 (K=64)
  v8f accM[4];
#pragma unroll
  for (int m = 0; m < 4; ++m) accM[m] = vzero();
#pragma unroll
  for (int ks = 0; ks < 2; ++ks) {
    const int k0 = ks * 32;
    v16bf Bv = load_b16(&tile[wv][0][0], 72, 0, k0, lane);
#pragma unroll
    for (int m = 0; m < 4; ++m) {
      v16bf Aa = load_a16(Wmid, 64, m * 16, k0, lane);
      accM[m] = wmma_bf16(Aa, Bv, accM[m]);
    }
  }
#pragma unroll
  for (int m = 0; m < 4; ++m)
#pragma unroll
    for (int r = 0; r < 8; ++r) {
      int ch = m * 16 + rhi + r;
      float v = accM[m][r] + Bmid[ch];
      v = v > 0.f ? v : 0.2f * v;
      tile[wv][px][ch] = f2bf(v);
    }
  // layer 3 (K=64) accumulates directly onto the skip result
#pragma unroll
  for (int ks = 0; ks < 2; ++ks) {
    const int k0 = ks * 32;
    v16bf Bv = load_b16(&tile[wv][0][0], 72, 0, k0, lane);
#pragma unroll
    for (int m = 0; m < 4; ++m) {
      v16bf Aa = load_a16(Wout, 64, m * 16, k0, lane);
      accS[m] = wmma_bf16(Aa, Bv, accS[m]);
    }
  }
  const float leakv = fminf(fmaxf(leakp[0], 0.001f), 1000.f);
#pragma unroll
  for (int m = 0; m < 2; ++m)
#pragma unroll
    for (int r = 0; r < 8; ++r) {
      int ch = m * 16 + rhi + r;              // 0..31
      float a = accS[m][r] + Bout[ch] + Bsk[ch];
      float g = accS[m + 2][r] + Bout[ch + 32] + Bsk[ch + 32];
      float on = a * (1.f / (1.f + __expf(-g)));   // hw v_exp_f32
      size_t off = ((size_t)b * 32 + ch) * 16384 + p0 + px;
      nxt[off] = prev[off] + leakv * on;
    }
}

// ---------------- final projection + clip ----------------
__global__ void k_final(const float* __restrict__ outf, const float* __restrict__ ow,
                        const float* __restrict__ ob,
                        float* __restrict__ dclip, float* __restrict__ draw) {
  int idx = blockIdx.x * 256 + threadIdx.x;
  if (idx >= 16 * 3 * 16384) return;
  int pix = idx & 16383;
  int o = (idx >> 14) % 3;
  int b = idx / (3 * 16384);
  float acc = ob[o];
#pragma unroll
  for (int i = 0; i < 32; ++i)
    acc = fmaf(outf[((size_t)b * 32 + i) * 16384 + pix], ow[o * 32 + i], acc);
  draw[idx] = acc;
  dclip[idx] = fminf(fmaxf(acc, -1.f), 1.f);
}

extern "C" void kernel_launch(void* const* d_in, const int* in_sizes, int n_in,
                              void* d_out, int out_size, void* d_ws, size_t ws_size,
                              hipStream_t stream) {
  (void)in_sizes; (void)n_in; (void)out_size; (void)ws_size;
  const float* lat     = (const float*)d_in[0];
  const float* leak    = (const float*)d_in[1];
  const float* waveb   = (const float*)d_in[2];
  const float* wt_w1   = (const float*)d_in[3];
  const float* wt_b1   = (const float*)d_in[4];
  const float* wt_w2   = (const float*)d_in[5];
  const float* wt_b2   = (const float*)d_in[6];
  const float* c2c_w1  = (const float*)d_in[7];
  const float* c2c_b1  = (const float*)d_in[8];
  const float* c2c_w2  = (const float*)d_in[9];
  const float* c2c_b2  = (const float*)d_in[10];
  const float* out_w   = (const float*)d_in[11];
  const float* out_b   = (const float*)d_in[12];

  // workspace carve (256B aligned)
  char* base = (char*)d_ws; size_t off = 0;
  auto alloc = [&](size_t bytes) -> void* {
    void* r = base + off;
    off = (off + bytes + 255) & ~(size_t)255;
    return r;
  };
  float*  h1    = (float*)alloc(16 * 512 * sizeof(float));
  float*  h2    = (float*)alloc(16 * 1024 * sizeof(float));
  float*  pw    = (float*)alloc((size_t)16 * 65568 * sizeof(float));
  float*  pp    = (float*)alloc((size_t)16 * 24832 * sizeof(float));
  bf16_t* ww    = (bf16_t*)alloc((size_t)16 * 65536 * sizeof(bf16_t));
  float*  wbb   = (float*)alloc(16 * 32 * sizeof(float));
  bf16_t* w_in  = (bf16_t*)alloc((size_t)16 * 8192 * sizeof(bf16_t));
  bf16_t* w_mid = (bf16_t*)alloc((size_t)16 * 4096 * sizeof(bf16_t));
  bf16_t* w_out = (bf16_t*)alloc((size_t)16 * 4096 * sizeof(bf16_t));
  bf16_t* w_sk  = (bf16_t*)alloc((size_t)16 * 8192 * sizeof(bf16_t));
  float*  b_in  = (float*)alloc(16 * 64 * sizeof(float));
  float*  b_mid = (float*)alloc(16 * 64 * sizeof(float));
  float*  b_out = (float*)alloc(16 * 64 * sizeof(float));
  float*  b_sk  = (float*)alloc(16 * 64 * sizeof(float));
  bf16_t* wbT   = (bf16_t*)alloc((size_t)16384 * 2048 * sizeof(bf16_t));
  bf16_t* xin   = (bf16_t*)alloc((size_t)16 * 16384 * 128 * sizeof(bf16_t));
  bf16_t* graw  = (bf16_t*)alloc((size_t)16 * 64 * 16384 * sizeof(bf16_t));
  float*  wstat = (float*)alloc(16 * 128 * sizeof(float));
  float*  gstat = (float*)alloc(16 * 128 * sizeof(float));

  // d_out layout: [clip(16,3,128,128)][embs(13,16,32,128,128)][raw(16,3,128,128)]
  float* dclip = (float*)d_out;
  float* emb   = dclip + 786432;
  float* draw  = emb + (size_t)13 * 8388608;
  const size_t EMB = 8388608;

  // ---- prologue ----
  hipMemsetAsync(emb, 0, EMB * sizeof(float), stream);        // out0 = zeros
  hipMemsetAsync(wstat, 0, 16 * 128 * sizeof(float), stream);
  k_hyper1<<<96, 256, 0, stream>>>(lat, wt_w1, wt_b1, c2c_w1, c2c_b1, h1, h2);
  k_hyper2<<<(65568 + 24832 + 255) / 256, 256, 0, stream>>>(h1, wt_w2, wt_b2, h2, c2c_w2, c2c_b2, pw, pp);
  k_unpack_wave<<<(16 * 65568 + 255) / 256, 256, 0, stream>>>(pw, ww, wbb);
  k_unpack_c2c<<<(16 * 24832 + 255) / 256, 256, 0, stream>>>(pp, w_in, b_in, w_mid, b_mid,
                                                             w_out, b_out, w_sk, b_sk);
  k_transpose_wb<<<dim3(512, 64), 256, 0, stream>>>(waveb, wbT);
  k_wavegemm<<<dim3(128, 16), 256, 0, stream>>>(wbT, ww, wbb, xin, wstat);
  k_norm<<<8192, 256, 0, stream>>>(xin, wstat, 64);

  // ---- 12-step recurrence ----
  for (int s = 0; s < 12; ++s) {
    hipMemsetAsync(gstat, 0, 16 * 128 * sizeof(float), stream);
    k_sobel<<<dim3(64, 32, 16), 256, 0, stream>>>(emb + (size_t)s * EMB, graw, gstat);
    k_packg<<<dim3(512, 2, 16), 256, 0, stream>>>(graw, gstat, xin);
    k_chain<<<dim3(128, 16), 256, 0, stream>>>(xin, w_in, b_in, w_mid, b_mid, w_out, b_out,
                                               w_sk, b_sk, leak,
                                               emb + (size_t)s * EMB, emb + (size_t)(s + 1) * EMB);
  }

  // ---- epilogue ----
  k_final<<<(786432 + 255) / 256, 256, 0, stream>>>(emb + (size_t)12 * EMB, out_w, out_b, dclip, draw);
}
